// IBSelectorModel_71829033058545
// MI455X (gfx1250) — matrix operand order repaired and checked by reading, not compiled, not run on hardware
//
#include <hip/hip_runtime.h>
#include <hip/hip_bf16.h>
#include <math.h>

typedef __bf16 bhalf;
typedef __attribute__((ext_vector_type(16))) bhalf v16b;
typedef __attribute__((ext_vector_type(8)))  bhalf v8b;
typedef __attribute__((ext_vector_type(8)))  float v8f;
typedef __attribute__((ext_vector_type(4)))  unsigned int u32x4;
typedef __attribute__((ext_vector_type(8)))  int i32x8;
typedef __attribute__((ext_vector_type(4)))  int i32x4;

#define B_    512
#define T_    200
#define D_    128
#define NH_   2
#define DH_   64
#define DFF_  2048
#define HID_  256
#define V_    100001
#define K_SEL 10
#define BT_   (B_*T_)
#define TP_   224           // T padded to 7*32 for WMMA K loops
#define LDT   40            // LDS row stride (elements): 32 data + 8 pad (TDM pads this)

// ---------------------------------------------------------------------------
// TDM 2D tile load: 128 rows x 32 bf16 tile, row stride = row_stride elems,
// written to LDS with a 16B pad every 64B -> effective LDS stride 80B (LDT=40).
// D# layout per CDNA5 ISA (groups 0..3 + cpol).
// Rows/cols past tensor_d1/tensor_d0 read as zero (hardware OOB handling).
// Toolchain uses the 6-arg builtin: (g0 u32x4, g1 i32x8, g2 i32x4, g3 i32x4,
// extra i32x8, cpol i32).
// ---------------------------------------------------------------------------
__device__ __forceinline__ void tdm_load_tile_128x32(unsigned lds_off, const bhalf* gptr,
                                                     int tensor_d0, int tensor_d1,
                                                     long long row_stride)
{
  unsigned long long ga = (unsigned long long)(const void*)gptr;
  u32x4 g0;
  g0[0] = 1u;                                          // count=1 (valid user D#)
  g0[1] = lds_off;                                     // lds_addr (bytes)
  g0[2] = (unsigned)(ga & 0xFFFFFFFFu);                // global_addr[31:0]
  g0[3] = (unsigned)((ga >> 32) & 0x01FFFFFFu)         // global_addr[56:32]
        | (2u << 30);                                  // type = 2 ("image")
  unsigned w0 = (1u << 16)                             // data_size = 2 bytes
              | (1u << 20)                             // pad_enable
              | (3u << 22)                             // pad_interval: every 64B (16 DW)
              | (3u << 25);                            // pad_amount: 4 DW = 16B
  unsigned td0 = (unsigned)tensor_d0;
  unsigned td1 = (unsigned)tensor_d1;
  unsigned long long s0 = (unsigned long long)row_stride;   // dim0 stride (elements)
  i32x8 g1;
  g1[0] = (int)w0;
  g1[1] = (int)((td0 & 0xFFFFu) << 16);                // [47:32]=atomic_barrier=0, [63:48]=td0.lo
  g1[2] = (int)((td0 >> 16) | ((td1 & 0xFFFFu) << 16));// [79:64]=td0.hi, [95:80]=td1.lo
  g1[3] = (int)(((td1 >> 16) & 0xFFFFu) | (32u << 16));// [111:96]=td1.hi, [127:112]=tile_dim0=32
  g1[4] = (int)128;                                    // [143:128]=tile_dim1=128, tile_dim2=0
  g1[5] = (int)(s0 & 0xFFFFFFFFull);                   // tensor_dim0_stride[31:0]
  g1[6] = (int)((s0 >> 32) & 0xFFFFull);               // stride[47:32]; dim1_stride.lo=0
  g1[7] = 0;
  i32x4 z4 = {0, 0, 0, 0};
  i32x8 z8 = {0, 0, 0, 0, 0, 0, 0, 0};
  __builtin_amdgcn_tensor_load_to_lds(g0, g1, z4, z4, z8, 0);
}

// ---------------------------------------------------------------------------
// Generic bf16 WMMA GEMM:  C[M,N] = A[M,K] * W[N,K]^T + bias[N]
// BM=BN=128, BK=32, 256 threads = 8 waves (wave32), each wave does 32x64.
// TDM double-buffered LDS staging: wave 0 issues tensor_load_to_lds for tile
// k+1 while all waves run WMMA on tile k; sync = s_wait_tensorcnt + barrier.
// M multiple of 128, K multiple of 32; N bounds-checked (TDM zero-fills OOB).
// ---------------------------------------------------------------------------
template<bool RELU>
__global__ void gemm_bf16_wmma(const bhalf* __restrict__ A,
                               const bhalf* __restrict__ W,
                               const float* __restrict__ bias,
                               float* __restrict__ Cf,
                               bhalf* __restrict__ Cb,
                               int M, int N, int K)
{
  __shared__ bhalf sA[2][128 * LDT];
  __shared__ bhalf sB[2][128 * LDT];
  const int tid  = threadIdx.x;
  const int lane = tid & 31;
  const int wave = tid >> 5;
  const int mBase = blockIdx.y * 128;
  const int nBase = blockIdx.x * 128;
  const int wm = (wave >> 1) * 32;   // 4 waves along M
  const int wn = (wave & 1) * 64;    // 2 waves along N

  const int mfrag = lane & 15;
  const int hig   = lane >> 4;
  const int kb8   = hig * 8;
  const int kb16  = hig * 16;

  const unsigned sAoff[2] = { (unsigned)(size_t)&sA[0][0], (unsigned)(size_t)&sA[1][0] };
  const unsigned sBoff[2] = { (unsigned)(size_t)&sB[0][0], (unsigned)(size_t)&sB[1][0] };

  v8f acc[2][4];
  #pragma unroll
  for (int i = 0; i < 2; ++i)
    #pragma unroll
    for (int j = 0; j < 4; ++j) acc[i][j] = (v8f){0.f,0.f,0.f,0.f,0.f,0.f,0.f,0.f};

  const int KT = K >> 5;             // number of 32-wide K tiles

  // prologue: DMA tile 0 into buffer 0
  if (wave == 0) {
    tdm_load_tile_128x32(sAoff[0], A + (size_t)mBase * K, K, M - mBase, K);
    tdm_load_tile_128x32(sBoff[0], W + (size_t)nBase * K, K, N - nBase, K);
    __builtin_amdgcn_s_wait_tensorcnt(0);
  }
  __syncthreads();

  for (int kt = 0; kt < KT; ++kt) {
    const int cur = kt & 1;
    // DMA next tile into the other buffer while we compute on this one
    if (kt + 1 < KT && wave == 0) {
      const int k1 = (kt + 1) << 5;
      tdm_load_tile_128x32(sAoff[cur ^ 1], A + (size_t)mBase * K + k1, K - k1, M - mBase, K);
      tdm_load_tile_128x32(sBoff[cur ^ 1], W + (size_t)nBase * K + k1, K - k1, N - nBase, K);
    }

    // ---- load fragments from LDS and issue WMMA
    v16b af[2];
    #pragma unroll
    for (int i = 0; i < 2; ++i) {
      const bhalf* p = &sA[cur][(wm + i * 16 + mfrag) * LDT + kb16];
      v8b lo = *(const v8b*)p;
      v8b hi = *(const v8b*)(p + 8);
      #pragma unroll
      for (int e = 0; e < 8; ++e) { af[i][e] = lo[e]; af[i][8 + e] = hi[e]; }
    }
    v16b bw[4];
    #pragma unroll
    for (int j = 0; j < 4; ++j)
      bw[j] = *(const v16b*)&sB[cur][(wn + j * 16 + mfrag) * LDT + kb16];

    #pragma unroll
    for (int i = 0; i < 2; ++i)
      #pragma unroll
      for (int j = 0; j < 4; ++j)
        acc[i][j] = __builtin_amdgcn_wmma_f32_16x16x32_bf16(
            false, af[i], false, bw[j], (short)0, acc[i][j], false, false);

    // wave 0 waits for the in-flight DMA, then everyone syncs before swapping
    if (wave == 0) __builtin_amdgcn_s_wait_tensorcnt(0);
    __syncthreads();
  }

  // ---- epilogue: bias (+ReLU), store f32 and/or bf16
  #pragma unroll
  for (int i = 0; i < 2; ++i) {
    #pragma unroll
    for (int j = 0; j < 4; ++j) {
      int col = nBase + wn + j * 16 + mfrag;
      if (col >= N) continue;
      float bv = bias ? bias[col] : 0.0f;
      #pragma unroll
      for (int vv = 0; vv < 8; ++vv) {
        int row = mBase + wm + i * 16 + vv + 8 * hig;
        float val = acc[i][j][vv] + bv;
        if (RELU) val = fmaxf(val, 0.0f);
        size_t idx = (size_t)row * N + col;
        if (Cf) Cf[idx] = val;
        if (Cb) Cb[idx] = (bhalf)val;
      }
    }
  }
}

// ---------------------------------------------------------------------------
// Fused attention: one workgroup per (b,h). qkv is bf16 [B*T, 384].
// ---------------------------------------------------------------------------
#define ATT_SMEM ((3 * TP_ * DH_ + 8 * 16 * TP_) * 2)

__global__ void attention_kernel(const bhalf* __restrict__ qkv, bhalf* __restrict__ ob)
{
  extern __shared__ char smem[];
  bhalf* sQ  = (bhalf*)smem;          // [TP_][64]
  bhalf* sK  = sQ  + TP_ * DH_;       // [TP_][64]
  bhalf* sVt = sK  + TP_ * DH_;       // [64][TP_]  (V transposed)
  bhalf* sPb = sVt + DH_ * TP_;       // 8 waves x [16][TP_]

  const int tid  = threadIdx.x;
  const int lane = tid & 31;
  const int wave = tid >> 5;
  const int b = blockIdx.x >> 1;
  const int h = blockIdx.x & 1;

  for (int idx = tid; idx < TP_ * DH_; idx += 256) {
    int t = idx >> 6, d = idx & 63;
    bhalf q = (bhalf)0.0f, k = (bhalf)0.0f, vv = (bhalf)0.0f;
    if (t < T_) {
      size_t base = ((size_t)(b * T_ + t)) * (3 * D_) + h * DH_ + d;
      q  = qkv[base];
      k  = qkv[base + D_];
      vv = qkv[base + 2 * D_];
    }
    sQ[idx] = q; sK[idx] = k; sVt[d * TP_ + t] = vv;
  }
  __syncthreads();

  const int mfrag = lane & 15;
  const int hig   = lane >> 4;
  const int kb8   = hig * 8;
  const int kb16  = hig * 16;
  bhalf* myP = sPb + wave * 16 * TP_;

  for (int qt = wave; qt < 14; qt += 8) {
    const int qr = qt * 16;
    // Q fragments (A layout), d-chunks of 32
    v16b aQ[2];
    #pragma unroll
    for (int c = 0; c < 2; ++c) {
      const bhalf* p = &sQ[(qr + mfrag) * DH_ + c * 32 + kb16];
      v8b lo = *(const v8b*)p;
      v8b hi = *(const v8b*)(p + 8);
      #pragma unroll
      for (int e = 0; e < 8; ++e) { aQ[c][e] = lo[e]; aQ[c][8 + e] = hi[e]; }
    }
    // S = Q K^T  (14 column tiles of 16)
    v8f sc[14];
    #pragma unroll
    for (int nt = 0; nt < 14; ++nt) {
      v8f s = (v8f){0.f,0.f,0.f,0.f,0.f,0.f,0.f,0.f};
      #pragma unroll
      for (int c = 0; c < 2; ++c) {
        v16b bk = *(const v16b*)&sK[(nt * 16 + mfrag) * DH_ + c * 32 + kb16];
        s = __builtin_amdgcn_wmma_f32_16x16x32_bf16(false, aQ[c], false, bk, (short)0, s, false, false);
      }
      sc[nt] = s;
    }
    // scale + mask + row-wise softmax (C layout: lanes 0-15 = cols of a row)
    float lm[8], ls[8];
    #pragma unroll
    for (int vv = 0; vv < 8; ++vv) lm[vv] = -1e30f;
    #pragma unroll
    for (int nt = 0; nt < 14; ++nt) {
      int col = nt * 16 + mfrag;
      #pragma unroll
      for (int vv = 0; vv < 8; ++vv) {
        float x = sc[nt][vv] * 0.125f;   // 1/sqrt(64)
        if (col >= T_) x = -1e30f;
        sc[nt][vv] = x;
        lm[vv] = fmaxf(lm[vv], x);
      }
    }
    #pragma unroll
    for (int vv = 0; vv < 8; ++vv) {
      #pragma unroll
      for (int o = 8; o >= 1; o >>= 1) lm[vv] = fmaxf(lm[vv], __shfl_xor(lm[vv], o, 32));
      ls[vv] = 0.0f;
    }
    #pragma unroll
    for (int nt = 0; nt < 14; ++nt)
      #pragma unroll
      for (int vv = 0; vv < 8; ++vv) {
        float p = __expf(sc[nt][vv] - lm[vv]);
        sc[nt][vv] = p; ls[vv] += p;
      }
    #pragma unroll
    for (int vv = 0; vv < 8; ++vv) {
      #pragma unroll
      for (int o = 8; o >= 1; o >>= 1) ls[vv] += __shfl_xor(ls[vv], o, 32);
      ls[vv] = 1.0f / ls[vv];
    }
    // write P (bf16) into per-wave LDS tile, row-major [16][TP_]
    #pragma unroll
    for (int nt = 0; nt < 14; ++nt)
      #pragma unroll
      for (int vv = 0; vv < 8; ++vv)
        myP[(vv + 8 * hig) * TP_ + nt * 16 + mfrag] = (bhalf)(sc[nt][vv] * ls[vv]);

    // O = P V   (K loop over padded T, 7 chunks of 32; N = DH in 4 tiles)
    #pragma unroll
    for (int dt = 0; dt < 4; ++dt) {
      v8f accO = (v8f){0.f,0.f,0.f,0.f,0.f,0.f,0.f,0.f};
      #pragma unroll
      for (int kc = 0; kc < 7; ++kc) {
        v16b aP;
        const bhalf* p = &myP[mfrag * TP_ + kc * 32 + kb16];
        v8b lo = *(const v8b*)p;
        v8b hi = *(const v8b*)(p + 8);
        #pragma unroll
        for (int e = 0; e < 8; ++e) { aP[e] = lo[e]; aP[8 + e] = hi[e]; }
        v16b bv = *(const v16b*)&sVt[(dt * 16 + mfrag) * TP_ + kc * 32 + kb16];
        accO = __builtin_amdgcn_wmma_f32_16x16x32_bf16(false, aP, false, bv, (short)0, accO, false, false);
      }
      #pragma unroll
      for (int vv = 0; vv < 8; ++vv) {
        int t = qr + vv + 8 * hig;
        if (t < T_)
          ob[((size_t)(b * T_ + t)) * D_ + h * DH_ + dt * 16 + mfrag] = (bhalf)accO[vv];
      }
    }
  }
}

// ---------------------------------------------------------------------------
// Elementwise / reduction kernels
// ---------------------------------------------------------------------------
__global__ void cvt_bf16(const float* __restrict__ src, bhalf* __restrict__ dst, int n)
{
  int i = blockIdx.x * 256 + threadIdx.x;
  if (i < n) dst[i] = (bhalf)src[i];
}

__global__ void embed_kernel(const int* __restrict__ ids, const float* __restrict__ emb,
                             const float* __restrict__ pos, float* __restrict__ xf,
                             bhalf* __restrict__ xb)
{
  size_t i = (size_t)blockIdx.x * 256 + threadIdx.x;
  int token = (int)(i / D_);
  int d = (int)(i % D_);
  int t = token % T_;
  int id = ids[token];
  float val = emb[(size_t)id * D_ + d] + pos[t * D_ + d];
  xf[i] = val;
  xb[i] = (bhalf)val;
}

// residual + LayerNorm, one wave per token (D=128 -> 4 elems/lane)
__global__ void add_ln_kernel(const float* __restrict__ xin, const float* __restrict__ pin,
                              const float* __restrict__ gamma, const float* __restrict__ beta,
                              float* __restrict__ xout, bhalf* __restrict__ xbout)
{
  const int lane = threadIdx.x & 31;
  const int wave = threadIdx.x >> 5;
  const size_t token = (size_t)blockIdx.x * 8 + wave;
  const float* xi = xin + token * D_;
  const float* pi = pin + token * D_;
  const int d0 = lane * 4;
  float v4[4];
  #pragma unroll
  for (int e = 0; e < 4; ++e) v4[e] = xi[d0 + e] + pi[d0 + e];
  float s = v4[0] + v4[1] + v4[2] + v4[3];
  #pragma unroll
  for (int o = 16; o >= 1; o >>= 1) s += __shfl_xor(s, o, 32);
  float mean = s * (1.0f / D_);
  float q = 0.0f;
  #pragma unroll
  for (int e = 0; e < 4; ++e) { float dd = v4[e] - mean; q += dd * dd; }
  #pragma unroll
  for (int o = 16; o >= 1; o >>= 1) q += __shfl_xor(q, o, 32);
  float inv = rsqrtf(q * (1.0f / D_) + 1e-5f);
  #pragma unroll
  for (int e = 0; e < 4; ++e) {
    float y = (v4[e] - mean) * inv * gamma[d0 + e] + beta[d0 + e];
    xout[token * D_ + d0 + e] = y;
    xbout[token * D_ + d0 + e] = (bhalf)y;
  }
}

// selector logits + gumbel softmax + top-K hard mask + KL; one block per b
__global__ void selector_kernel(const float* __restrict__ H, const float* __restrict__ u,
                                const float* __restrict__ sw, const float* __restrict__ sb,
                                float* __restrict__ mask, float* __restrict__ klacc)
{
  __shared__ float zbuf[T_], softv[T_], work[T_], hard[T_];
  __shared__ float red[256];
  __shared__ int   ridx[256];
  const int tid = threadIdx.x;
  const int b = blockIdx.x;

  if (tid < T_) {
    const float* hrow = H + ((size_t)b * T_ + tid) * D_;
    float acc = 0.0f;
    for (int d = 0; d < D_; ++d) acc += hrow[d] * sw[d];
    float logit = acc + sb[0];
    float uu = u[(size_t)b * T_ + tid];
    float g = -logf(-logf(uu + 1e-9f) + 1e-9f);
    zbuf[tid] = (logit + g) * 2.0f;      // / tau, tau = 0.5
    hard[tid] = 0.0f;
  }
  __syncthreads();

  red[tid] = (tid < T_) ? zbuf[tid] : -1e30f;
  __syncthreads();
  for (int s2 = 128; s2 >= 1; s2 >>= 1) {
    if (tid < s2) red[tid] = fmaxf(red[tid], red[tid + s2]);
    __syncthreads();
  }
  float zmax = red[0];
  __syncthreads();
  float ex = (tid < T_) ? expf(zbuf[tid] - zmax) : 0.0f;
  red[tid] = ex;
  __syncthreads();
  for (int s2 = 128; s2 >= 1; s2 >>= 1) {
    if (tid < s2) red[tid] += red[tid + s2];
    __syncthreads();
  }
  float inv = 1.0f / red[0];
  __syncthreads();
  if (tid < T_) { float p = ex * inv; softv[tid] = p; work[tid] = p; }
  __syncthreads();

  // KL term: sum soft*(log(soft+1e-8)+log(T))
  red[tid] = (tid < T_) ? softv[tid] * (logf(softv[tid] + 1e-8f) + logf((float)T_)) : 0.0f;
  __syncthreads();
  for (int s2 = 128; s2 >= 1; s2 >>= 1) {
    if (tid < s2) red[tid] += red[tid + s2];
    __syncthreads();
  }
  if (tid == 0) atomicAdd(klacc, red[0]);
  __syncthreads();

  // iterative top-K argmax
  for (int it = 0; it < K_SEL; ++it) {
    red[tid]  = (tid < T_) ? work[tid] : -1e30f;
    ridx[tid] = tid;
    __syncthreads();
    for (int s2 = 128; s2 >= 1; s2 >>= 1) {
      if (tid < s2 && red[tid + s2] > red[tid]) {
        red[tid] = red[tid + s2]; ridx[tid] = ridx[tid + s2];
      }
      __syncthreads();
    }
    if (tid == 0) { hard[ridx[0]] = 1.0f; work[ridx[0]] = -1e30f; }
    __syncthreads();
  }
  if (tid < T_)
    mask[(size_t)b * T_ + tid] = (hard[tid] - softv[tid]) + softv[tid];
}

// masked mean pool: one block per b, 128 threads (one per d)
__global__ void pool_kernel(const float* __restrict__ H, const float* __restrict__ mask,
                            float* __restrict__ hs, bhalf* __restrict__ hsb)
{
  const int b = blockIdx.x, d = threadIdx.x;
  float acc = 0.0f;
  for (int t = 0; t < T_; ++t)
    acc += mask[(size_t)b * T_ + t] * H[((size_t)b * T_ + t) * D_ + d];
  acc *= (1.0f / K_SEL);
  hs[(size_t)b * D_ + d] = acc;
  hsb[(size_t)b * D_ + d] = (bhalf)acc;
}

__global__ void zero_kernel(float* p) { p[0] = 0.0f; }
__global__ void finalize_kl(const float* acc, float* out) { out[0] = acc[0] / (float)B_; }

// ---------------------------------------------------------------------------
extern "C" void kernel_launch(void* const* d_in, const int* in_sizes, int n_in,
                              void* d_out, int out_size, void* d_ws, size_t ws_size,
                              hipStream_t stream)
{
  const int*   input_ids = (const int*)d_in[0];
  const float* u        = (const float*)d_in[1];
  const float* item_emb = (const float*)d_in[2];
  const float* pos_emb  = (const float*)d_in[3];
  const float* Wqkv     = (const float*)d_in[4];
  const float* bqkv     = (const float*)d_in[5];
  const float* Wo       = (const float*)d_in[6];
  const float* bo       = (const float*)d_in[7];
  const float* ln1_s    = (const float*)d_in[8];
  const float* ln1_b    = (const float*)d_in[9];
  const float* W1       = (const float*)d_in[10];
  const float* b1       = (const float*)d_in[11];
  const float* W2       = (const float*)d_in[12];
  const float* b2       = (const float*)d_in[13];
  const float* ln2_s    = (const float*)d_in[14];
  const float* ln2_b    = (const float*)d_in[15];
  const float* sel_w    = (const float*)d_in[16];
  const float* sel_b    = (const float*)d_in[17];
  const float* p1_w     = (const float*)d_in[18];
  const float* p1_bias  = (const float*)d_in[19];
  const float* p2_w     = (const float*)d_in[20];
  const float* p2_bias  = (const float*)d_in[21];
  float* out = (float*)d_out;
  (void)in_sizes; (void)n_in; (void)out_size; (void)ws_size;

  char* ws = (char*)d_ws;
  size_t off = 0;
  auto alloc = [&](size_t bytes) -> char* {
    char* p = ws + off;
    off += (bytes + 255) & ~(size_t)255;
    return p;
  };
  float* xf    = (float*)alloc((size_t)BT_ * D_ * 4);
  bhalf* xb    = (bhalf*)alloc((size_t)BT_ * D_ * 2);
  bhalf* qkvb  = (bhalf*)alloc((size_t)BT_ * 3 * D_ * 2);
  bhalf* ob    = (bhalf*)alloc((size_t)BT_ * D_ * 2);
  float* pf    = (float*)alloc((size_t)BT_ * D_ * 4);
  bhalf* hb    = (bhalf*)alloc((size_t)12800 * DFF_ * 2);
  bhalf* wqkvb = (bhalf*)alloc((size_t)2 * 3 * D_ * D_ * 2);
  bhalf* wob   = (bhalf*)alloc((size_t)2 * D_ * D_ * 2);
  bhalf* w1b   = (bhalf*)alloc((size_t)2 * DFF_ * D_ * 2);
  bhalf* w2b   = (bhalf*)alloc((size_t)2 * D_ * DFF_ * 2);
  bhalf* p1wb  = (bhalf*)alloc((size_t)HID_ * D_ * 2);
  bhalf* p2wb  = (bhalf*)alloc((size_t)V_ * HID_ * 2);
  float* maskw = (float*)alloc((size_t)BT_ * 4);
  float* hsel  = (float*)alloc((size_t)B_ * D_ * 4);
  bhalf* hselb = (bhalf*)alloc((size_t)B_ * D_ * 2);
  bhalf* hidb  = (bhalf*)alloc((size_t)B_ * HID_ * 2);
  float* klacc = (float*)alloc(256);

  auto cvt = [&](const float* src, bhalf* dst, int n) {
    cvt_bf16<<<(n + 255) / 256, 256, 0, stream>>>(src, dst, n);
  };
  cvt(Wqkv, wqkvb, 2 * 3 * D_ * D_);
  cvt(Wo,   wob,   2 * D_ * D_);
  cvt(W1,   w1b,   2 * DFF_ * D_);
  cvt(W2,   w2b,   2 * D_ * DFF_);
  cvt(p1_w, p1wb,  HID_ * D_);
  cvt(p2_w, p2wb,  V_ * HID_);

  embed_kernel<<<(BT_ * D_) / 256, 256, 0, stream>>>(input_ids, item_emb, pos_emb, xf, xb);

  for (int l = 0; l < 2; ++l) {
    // QKV projection -> bf16 [BT, 384]
    gemm_bf16_wmma<false><<<dim3(3, BT_ / 128), 256, 0, stream>>>(
        xb, wqkvb + (size_t)l * 3 * D_ * D_, bqkv + l * 3 * D_,
        nullptr, qkvb, BT_, 3 * D_, D_);
    // fused attention -> bf16 [BT, 128]
    attention_kernel<<<B_ * NH_, 256, ATT_SMEM, stream>>>(qkvb, ob);
    // output projection -> f32 [BT, 128]
    gemm_bf16_wmma<false><<<dim3(1, BT_ / 128), 256, 0, stream>>>(
        ob, wob + (size_t)l * D_ * D_, bo + l * D_, pf, nullptr, BT_, D_, D_);
    // x = LN(x + proj)
    add_ln_kernel<<<BT_ / 8, 256, 0, stream>>>(xf, pf, ln1_s + l * D_, ln1_b + l * D_, xf, xb);
    // FFN in 8 M-chunks of 12800 rows
    for (int c = 0; c < 8; ++c) {
      size_t m0 = (size_t)c * 12800;
      gemm_bf16_wmma<true><<<dim3(DFF_ / 128, 100), 256, 0, stream>>>(
          xb + m0 * D_, w1b + (size_t)l * DFF_ * D_, b1 + l * DFF_,
          nullptr, hb, 12800, DFF_, D_);
      gemm_bf16_wmma<false><<<dim3(1, 100), 256, 0, stream>>>(
          hb, w2b + (size_t)l * D_ * DFF_, b2 + l * D_,
          pf + m0 * D_, nullptr, 12800, D_, DFF_);
    }
    add_ln_kernel<<<BT_ / 8, 256, 0, stream>>>(xf, pf, ln2_s + l * D_, ln2_b + l * D_, xf, xb);
  }

  zero_kernel<<<1, 1, 0, stream>>>(klacc);
  selector_kernel<<<B_, 256, 0, stream>>>(xf, u, sel_w, sel_b, maskw, klacc);
  pool_kernel<<<B_, 128, 0, stream>>>(xf, maskw, hsel, hselb);

  // hid = relu(h_sel @ p1^T + b)
  gemm_bf16_wmma<true><<<dim3(HID_ / 128, B_ / 128), 256, 0, stream>>>(
      hselb, p1wb, p1_bias, nullptr, hidb, B_, HID_, D_);
  // logits_pred = hid @ p2^T + b  -> f32 directly into d_out
  gemm_bf16_wmma<false><<<dim3((V_ + 127) / 128, B_ / 128), 256, 0, stream>>>(
      hidb, p2wb, p2_bias, out, nullptr, B_, V_, HID_);

  finalize_kl<<<1, 1, 0, stream>>>(klacc, out + (size_t)B_ * V_);
}
